// MultiHeadAttention_3831110828711
// MI455X (gfx1250) — matrix operand order repaired and checked by reading, not compiled, not run on hardware
//
#include <hip/hip_runtime.h>
#include <hip/hip_bf16.h>

// ---------------------------------------------------------------------------
// Multi-head attention, MI455X (gfx1250, wave32, WMMA f32<-f16).
// B=4, E=768, S=2048, H=12, D=64.  All GEMMs via v_wmma_f32_16x16x32_f16.
// GEMM kernels are register-blocked: one wave owns a 32x64 C block
// (2 A-fragments x 4 B-fragments -> 8 WMMAs per 32-wide K step).
// Fragment layouts follow CDNA5 ISA 7.12.2:
//   A 16x32 f16 : lane m=lane&15, koff=8*(lane>>4); elems 0..7 -> K=koff+j,
//                 elems 8..15 -> K=koff+16+(j-8).
//   B 32x16 f16 : lane n=lane&15, same K order as A (koff=8*(lane>>4)).
//   C 16x16 f32 : elem r -> row r+8*(lane>>4), col lane&15.
// ---------------------------------------------------------------------------

typedef _Float16 v16h __attribute__((ext_vector_type(16)));
typedef _Float16 v8h  __attribute__((ext_vector_type(8)));
typedef float    v8f  __attribute__((ext_vector_type(8)));

constexpr int B = 4, E = 768, S = 2048, H = 12, D = 64;

__device__ __forceinline__ v8f wmma_f16(v16h a, v16h b, v8f c) {
  return __builtin_amdgcn_wmma_f32_16x16x32_f16(false, a, false, b,
                                                (short)0, c, false, false);
}

// A fragment: 16x32 tile, row-major source with row stride ld (elements).
__device__ __forceinline__ v16h load_frag_a(const _Float16* base, int ld, int lane) {
  const int m = lane & 15;
  const int koff = (lane >> 4) * 8;
  const _Float16* p = base + (size_t)m * ld + koff;
  v8h lo = *(const v8h*)(p);       // K = koff .. koff+7
  v8h hi = *(const v8h*)(p + 16);  // K = koff+16 .. koff+23
  return __builtin_shufflevector(lo, hi, 0,1,2,3,4,5,6,7,8,9,10,11,12,13,14,15);
}

// B fragment: 32x16 tile given B^T row-major (rows = N, cols = K, stride ld).
__device__ __forceinline__ v16h load_frag_bt(const _Float16* base, int ld, int lane) {
  const int n = lane & 15;
  const int kb = (lane >> 4) * 8;
  const _Float16* p = base + (size_t)n * ld + kb;
  v8h lo = *(const v8h*)(p);       // K = kb .. kb+7
  v8h hi = *(const v8h*)(p + 16);  // K = kb+16 .. kb+23
  return __builtin_shufflevector(lo, hi, 0,1,2,3,4,5,6,7,8,9,10,11,12,13,14,15);
}

// --- layout / precision conversion -----------------------------------------

// hidden [B,E,S] f32 -> xh [B,S,E] f16
__global__ void k_xcast(const float* __restrict__ hs, _Float16* __restrict__ xh) {
  int idx = blockIdx.x * blockDim.x + threadIdx.x;
  if (idx >= B * E * S) return;
  int s = idx % S;
  int e = (idx / S) % E;
  int b = idx / (S * E);
  xh[((size_t)b * S + s) * E + e] = (_Float16)hs[idx];
}

// Wq/Wk/Wv [H,E,D] -> [H,D,E] f16 ; Wo [E,E] -> f16 copy.  (H*E*D == E*E)
__global__ void k_wcast(const float* __restrict__ Wq, const float* __restrict__ Wk,
                        const float* __restrict__ Wv, const float* __restrict__ Wo,
                        _Float16* __restrict__ Wqt, _Float16* __restrict__ Wkt,
                        _Float16* __restrict__ Wvt, _Float16* __restrict__ Woh) {
  int idx = blockIdx.x * blockDim.x + threadIdx.x;
  if (idx >= H * E * D) return;
  int which = blockIdx.y;
  if (which == 3) { Woh[idx] = (_Float16)Wo[idx]; return; }
  const float* W = (which == 0) ? Wq : (which == 1) ? Wk : Wv;
  _Float16* Wt   = (which == 0) ? Wqt : (which == 1) ? Wkt : Wvt;
  int d = idx % D;
  int e = (idx / D) % E;
  int h = idx / (D * E);
  Wt[((size_t)h * D + d) * E + e] = (_Float16)W[idx];
}

// --- QKV projection: [32,E] @ [E,64] block per wave per (b,h); z = Q/K/V ----

__global__ void __launch_bounds__(32) k_qkv(
    const _Float16* __restrict__ xh,
    const _Float16* __restrict__ Wqt, const _Float16* __restrict__ Wkt,
    const _Float16* __restrict__ Wvt,
    const float* __restrict__ bq, const float* __restrict__ bk,
    const float* __restrict__ bv,
    _Float16* __restrict__ Qh, _Float16* __restrict__ Kh,
    _Float16* __restrict__ Vt) {
  const int lane = threadIdx.x & 31;
  const int mt = blockIdx.x;       // S/32 tiles
  const int bh = blockIdx.y;
  const int b = bh / H, h = bh % H;
  const int z = blockIdx.z;
  const _Float16* Wt = (z == 0) ? Wqt : (z == 1) ? Wkt : Wvt;
  const float* bias  = (z == 0) ? bq  : (z == 1) ? bk  : bv;
  const int s0 = mt * 32;
  const _Float16* Ab = xh + ((size_t)b * S + s0) * E;
  const _Float16* Bb = Wt + (size_t)h * D * E;  // 4 N-tiles of 16 rows each
  v8f acc[2][4];
  const v8f zero = {};
#pragma unroll
  for (int mi = 0; mi < 2; ++mi)
#pragma unroll
    for (int n = 0; n < 4; ++n) acc[mi][n] = zero;

  for (int e0 = 0; e0 < E; e0 += 32) {
    v16h a0 = load_frag_a(Ab + e0, E, lane);
    v16h a1 = load_frag_a(Ab + (size_t)16 * E + e0, E, lane);
#pragma unroll
    for (int n = 0; n < 4; ++n) {
      v16h w = load_frag_bt(Bb + (size_t)(n * 16) * E + e0, E, lane);
      acc[0][n] = wmma_f16(a0, w, acc[0][n]);
      acc[1][n] = wmma_f16(a1, w, acc[1][n]);
    }
  }
  const int col = lane & 15, half = lane >> 4;
#pragma unroll
  for (int n = 0; n < 4; ++n) {
    const float bb = bias[h * D + n * 16 + col];
#pragma unroll
    for (int mi = 0; mi < 2; ++mi) {
#pragma unroll
      for (int r = 0; r < 8; ++r) {
        int row = s0 + mi * 16 + r + 8 * half;
        float v = acc[mi][n][r] + bb;
        if (z < 2) {  // Q/K: [BH,S,D]
          _Float16* dst = (z == 0) ? Qh : Kh;
          dst[((size_t)bh * S + row) * D + n * 16 + col] = (_Float16)v;
        } else {      // V stored transposed: [BH,D,S]
          Vt[((size_t)bh * D + n * 16 + col) * S + row] = (_Float16)v;
        }
      }
    }
  }
}

// --- flash attention: 16 queries / wave, 32-key blocks, online softmax ------

__global__ void __launch_bounds__(32) k_attn(
    const _Float16* __restrict__ Qh, const _Float16* __restrict__ Kh,
    const _Float16* __restrict__ Vt, _Float16* __restrict__ Ctxh) {
  __shared__ __align__(16) _Float16 Plds[16 * 32];  // one wave per block
  const int lane = threadIdx.x & 31;
  const int qb = blockIdx.x;   // S/16
  const int bh = blockIdx.y;
  const int b = bh / H, h = bh % H;
  const int half = lane >> 4, col = lane & 15;

  const _Float16* Qb = Qh + ((size_t)bh * S + qb * 16) * D;
  const v16h a0 = load_frag_a(Qb, D, lane);       // d = 0..31
  const v16h a1 = load_frag_a(Qb + 32, D, lane);  // d = 32..63

  float mr[8], lr[8];
  v8f ctx[4];
  const v8f zero = {};
#pragma unroll
  for (int r = 0; r < 8; ++r) { mr[r] = -1e30f; lr[r] = 0.0f; }
#pragma unroll
  for (int t = 0; t < 4; ++t) ctx[t] = zero;

  for (int kb = 0; kb < S; kb += 32) {
    const _Float16* Kb = Kh + ((size_t)bh * S + kb) * D;
    // scores: Q[16x64] @ K^T[64x32]  (no 1/sqrt(D) -- faithful to reference)
    v8f sc0 = zero, sc1 = zero;
    sc0 = wmma_f16(a0, load_frag_bt(Kb, D, lane), sc0);
    sc0 = wmma_f16(a1, load_frag_bt(Kb + 32, D, lane), sc0);
    sc1 = wmma_f16(a0, load_frag_bt(Kb + (size_t)16 * D, D, lane), sc1);
    sc1 = wmma_f16(a1, load_frag_bt(Kb + (size_t)16 * D + 32, D, lane), sc1);

    float p0[8], p1[8];
#pragma unroll
    for (int r = 0; r < 8; ++r) {
      float mx = fmaxf(sc0[r], sc1[r]);
      mx = fmaxf(mx, __shfl_xor(mx, 1, 32));
      mx = fmaxf(mx, __shfl_xor(mx, 2, 32));
      mx = fmaxf(mx, __shfl_xor(mx, 4, 32));
      mx = fmaxf(mx, __shfl_xor(mx, 8, 32));   // row lives inside a 16-lane half
      float mn = fmaxf(mr[r], mx);
      float scale = __expf(mr[r] - mn);
      float e0 = __expf(sc0[r] - mn);
      float e1 = __expf(sc1[r] - mn);
      float rs = e0 + e1;
      rs += __shfl_xor(rs, 1, 32);
      rs += __shfl_xor(rs, 2, 32);
      rs += __shfl_xor(rs, 4, 32);
      rs += __shfl_xor(rs, 8, 32);
      lr[r] = lr[r] * scale + rs;
      mr[r] = mn;
#pragma unroll
      for (int t = 0; t < 4; ++t) ctx[t][r] *= scale;
      p0[r] = e0; p1[r] = e1;
    }
    // P (C layout) -> LDS -> A-fragment layout
#pragma unroll
    for (int r = 0; r < 8; ++r) {
      int row = r + 8 * half;
      Plds[row * 32 + col]      = (_Float16)p0[r];
      Plds[row * 32 + 16 + col] = (_Float16)p1[r];
    }
    __syncthreads();
    v16h ap = load_frag_a(Plds, 32, lane);
    const _Float16* Vb = Vt + (size_t)bh * D * S + kb;  // [D,S], keys contiguous
#pragma unroll
    for (int t = 0; t < 4; ++t) {
      v16h bv = load_frag_bt(Vb + (size_t)(t * 16) * S, S, lane);
      ctx[t] = wmma_f16(ap, bv, ctx[t]);
    }
    __syncthreads();
  }
#pragma unroll
  for (int t = 0; t < 4; ++t) {
#pragma unroll
    for (int r = 0; r < 8; ++r) {
      int row = qb * 16 + r + 8 * half;
      float val = ctx[t][r] / lr[r];
      Ctxh[((size_t)b * S + row) * E + h * 64 + t * 16 + col] = (_Float16)val;
    }
  }
}

// --- output projection: ctx[S,E] @ Wo^T + bo, 32x64 block/wave, out [B,E,S] -

__global__ void __launch_bounds__(32) k_out(
    const _Float16* __restrict__ Ctxh, const _Float16* __restrict__ Woh,
    const float* __restrict__ bo, float* __restrict__ out) {
  const int lane = threadIdx.x & 31;
  const int mt = blockIdx.x % (S / 32);   // 64 row tiles
  const int nt = blockIdx.x / (S / 32);   // 12 col tiles of 64
  const int b = blockIdx.y;
  const _Float16* Ab = Ctxh + ((size_t)b * S + mt * 32) * E;
  const _Float16* Bb = Woh + (size_t)(nt * 64) * E;  // Wo row-major == B^T
  v8f acc[2][4];
  const v8f zero = {};
#pragma unroll
  for (int mi = 0; mi < 2; ++mi)
#pragma unroll
    for (int n = 0; n < 4; ++n) acc[mi][n] = zero;

  for (int e0 = 0; e0 < E; e0 += 32) {
    v16h a0 = load_frag_a(Ab + e0, E, lane);
    v16h a1 = load_frag_a(Ab + (size_t)16 * E + e0, E, lane);
#pragma unroll
    for (int n = 0; n < 4; ++n) {
      v16h w = load_frag_bt(Bb + (size_t)(n * 16) * E + e0, E, lane);
      acc[0][n] = wmma_f16(a0, w, acc[0][n]);
      acc[1][n] = wmma_f16(a1, w, acc[1][n]);
    }
  }
  const int col = lane & 15, half = lane >> 4;
#pragma unroll
  for (int n = 0; n < 4; ++n) {
    const float bb = bo[nt * 64 + n * 16 + col];
#pragma unroll
    for (int mi = 0; mi < 2; ++mi) {
#pragma unroll
      for (int r = 0; r < 8; ++r) {
        int row = mt * 32 + mi * 16 + r + 8 * half;  // s index
        out[(size_t)b * E * S + (size_t)(nt * 64 + n * 16 + col) * S + row] =
            acc[mi][n][r] + bb;
      }
    }
  }
}

// ---------------------------------------------------------------------------

extern "C" void kernel_launch(void* const* d_in, const int* in_sizes, int n_in,
                              void* d_out, int out_size, void* d_ws, size_t ws_size,
                              hipStream_t stream) {
  const float* hidden = (const float*)d_in[0];
  const float* Wq = (const float*)d_in[1];
  const float* bq = (const float*)d_in[2];
  const float* Wk = (const float*)d_in[3];
  const float* bk = (const float*)d_in[4];
  const float* Wv = (const float*)d_in[5];
  const float* bv = (const float*)d_in[6];
  const float* Wo = (const float*)d_in[7];
  const float* bo = (const float*)d_in[8];
  float* out = (float*)d_out;

  const size_t nxh  = (size_t)B * S * E;      // 6,291,456
  const size_t nwt  = (size_t)H * D * E;      //   589,824 (== E*E)
  const size_t nqkv = (size_t)B * H * S * D;  // 6,291,456

  _Float16* p    = (_Float16*)d_ws;           // total ~67.6 MB of f16 scratch
  _Float16* xh   = p; p += nxh;
  _Float16* Wqt  = p; p += nwt;
  _Float16* Wkt  = p; p += nwt;
  _Float16* Wvt  = p; p += nwt;
  _Float16* Woh  = p; p += nwt;
  _Float16* Qh   = p; p += nqkv;
  _Float16* Kh   = p; p += nqkv;
  _Float16* Vt   = p; p += nqkv;
  _Float16* Ctxh = p; p += nxh;

  k_xcast<<<dim3((B * E * S + 255) / 256), 256, 0, stream>>>(hidden, xh);
  k_wcast<<<dim3((int)((nwt + 255) / 256), 4), 256, 0, stream>>>(
      Wq, Wk, Wv, Wo, Wqt, Wkt, Wvt, Woh);
  k_qkv<<<dim3(S / 32, B * H, 3), 32, 0, stream>>>(
      xh, Wqt, Wkt, Wvt, bq, bk, bv, Qh, Kh, Vt);
  k_attn<<<dim3(S / 16, B * H), 32, 0, stream>>>(Qh, Kh, Vt, Ctxh);
  k_out<<<dim3((S / 32) * (E / 64), B), 32, 0, stream>>>(Ctxh, Woh, bo, out);
}
